// LorentzMLA_3891240370243
// MI455X (gfx1250) — matrix operand order repaired and checked by reading, not compiled, not run on hardware
//
#include <hip/hip_runtime.h>
#include <stdint.h>

// ---------------------------------------------------------------------------
// Lorentz-MLA forward for MI455X (gfx1250, wave32, WMMA bf16 16x16x32).
// GEMMs (projections, QK^T, P*V) run on v_wmma_f32_16x16x32_bf16 with f32
// accumulation.  The projection GEMMs stage A/B panels into LDS with
// gfx1250 async-to-LDS DMA (global_load_async_to_lds_b128 + s_wait_asynccnt).
// ---------------------------------------------------------------------------

typedef __bf16 bf16;
typedef __attribute__((ext_vector_type(16))) __bf16 v16bf;
typedef __attribute__((ext_vector_type(8)))  __bf16 v8bf;
typedef __attribute__((ext_vector_type(8)))  float  v8f;

#define S_LEN   2048
#define DIM_    2048
#define H_HEADS 16
#define SPAD    2080      // S + 32 slack rows so boundary key tiles never read OOB
#define DQK_P   224       // 193 (projected qk dim) padded to 7*32
#define DV      128       // projected value dim (VHD)
#define KVRANK  512
#define KP_KV   544       // 513 padded to 17*32
#define N_Q     3072      // H * 192
#define N_KVA   576       // KV_RANK + 64 (real)
#define N_KVA_P 640       // padded to multiple of 128 for the block-tiled GEMM
#define N_KVB   4080      // H * 255 (real)
#define N_KVB_P 4096      // padded to multiple of 128

#define BM 128
#define BN 128
#define BK 32

// ---------------------------------------------------------------------------
// helpers
// ---------------------------------------------------------------------------
__device__ __forceinline__ bf16 f2bf(float f) {
  union { float f; uint32_t u; } a; a.f = f;
  uint32_t r = a.u + 0x7FFFu + ((a.u >> 16) & 1u);   // round-to-nearest-even
  union { uint16_t u; bf16 b; } o; o.u = (uint16_t)(r >> 16);
  return o.b;
}

__device__ __forceinline__ float rsum32(float v) {
#pragma unroll
  for (int m = 1; m < 32; m <<= 1) v += __shfl_xor(v, m, 32);
  return v;
}
__device__ __forceinline__ float rsum16(float v) {
#pragma unroll
  for (int m = 1; m < 16; m <<= 1) v += __shfl_xor(v, m, 16);
  return v;
}
__device__ __forceinline__ float rmax16(float v) {
#pragma unroll
  for (int m = 1; m < 16; m <<= 1) v = fmaxf(v, __shfl_xor(v, m, 16));
  return v;
}

__device__ __forceinline__ v8f wmma_bf16(v16bf a, v16bf b, v8f c) {
  return __builtin_amdgcn_wmma_f32_16x16x32_bf16(false, a, false, b, (short)0, c,
                                                 false, false);
}

// A fragment (16x32 bf16, row-major MxK source). ISA layout: lane = half*16+m,
// elements 0..7 hold K = half*8 + 0..7, elements 8..15 hold K = 16+half*8+0..7
// => two contiguous 16B loads per lane.
__device__ __forceinline__ v16bf load_fragA(const bf16* __restrict__ base, int ldk,
                                            int row0, int kk, int lane) {
  int m = lane & 15, hh = lane >> 4;
  const bf16* p = base + (size_t)(row0 + m) * ldk + kk + hh * 8;
  v8bf lo = *(const v8bf*)p;
  v8bf hi = *(const v8bf*)(p + 16);
  v16bf r;
#pragma unroll
  for (int i = 0; i < 8; ++i) { r[i] = lo[i]; r[i + 8] = hi[i]; }
  return r;
}

// B fragment (32x16 bf16) from a column-major-equivalent source: bt is NxK
// row-major (i.e. B transposed). Elements 0..7 hold K = half*8 + 0..7,
// elements 8..15 hold K = 16+half*8+0..7 => two 16B loads per lane.
__device__ __forceinline__ v16bf load_fragB(const bf16* __restrict__ bt, int ldk,
                                            int col0, int kk, int lane) {
  int n = lane & 15, hh = lane >> 4;
  const bf16* p = bt + (size_t)(col0 + n) * ldk + kk + hh * 8;
  v8bf lo = *(const v8bf*)p;
  v8bf hi = *(const v8bf*)(p + 16);
  v16bf r;
#pragma unroll
  for (int i = 0; i < 8; ++i) { r[i] = lo[i]; r[i + 8] = hi[i]; }
  return r;
}

// gfx1250 async DMA: global memory -> LDS, 16 bytes per lane, ASYNCcnt-tracked
__device__ __forceinline__ void async_g2l_b128(uint32_t lds_byte_addr,
                                               const void* gaddr) {
  asm volatile("global_load_async_to_lds_b128 %0, %1, off"
               :: "v"(lds_byte_addr), "v"(gaddr) : "memory");
}
__device__ __forceinline__ void wait_async0() {
  asm volatile("s_wait_asynccnt 0" ::: "memory");
}

// ---------------------------------------------------------------------------
// conversion kernels (fp32 -> bf16, with zero padding; optional transpose)
// ---------------------------------------------------------------------------
__global__ void cvt_pad(const float* __restrict__ src, bf16* __restrict__ dst,
                        int R, int C, int Cp) {
  size_t idx = (size_t)blockIdx.x * blockDim.x + threadIdx.x;
  if (idx >= (size_t)R * Cp) return;
  int r = (int)(idx / Cp), c = (int)(idx % Cp);
  dst[idx] = f2bf(c < C ? src[(size_t)r * C + c] : 0.0f);
}

__global__ void cvt_tpad(const float* __restrict__ src, bf16* __restrict__ dst,
                         int K, int N, int Np, int Kp) {
  size_t idx = (size_t)blockIdx.x * blockDim.x + threadIdx.x;
  if (idx >= (size_t)Np * Kp) return;
  int n = (int)(idx / Kp), k = (int)(idx % Kp);
  float v = (n < N && k < K) ? src[(size_t)k * N + n] : 0.0f;
  dst[idx] = f2bf(v);
}

// ---------------------------------------------------------------------------
// block-tiled WMMA GEMM: C(MxN,f32) = A(MxKp,bf16 row-major) * Bt(NxKp,bf16)^T
// 256 threads compute a 128x128 C tile; per 32-wide K step the 128x32 A and
// Bt panels are staged into LDS via async DMA; each wave owns a 32x64 subtile
// (2 A frags x 4 B frags -> 8 WMMAs per K step).
// Requires: M % 128 == 0, N % 128 == 0, Kp % 32 == 0.
// ---------------------------------------------------------------------------
__global__ __launch_bounds__(256, 1) void gemm_bf16_kernel(
    const bf16* __restrict__ A, const bf16* __restrict__ Bt,
    float* __restrict__ C, int M, int N, int Kp) {
  __shared__ __align__(32) bf16 As[BM * BK];
  __shared__ __align__(32) bf16 Bs[BN * BK];
  (void)M;
  int tid = threadIdx.x;
  int lane = tid & 31;
  int wv = tid >> 5;
  int tilesN = N / BN;
  int bM = blockIdx.x / tilesN, bN = blockIdx.x - bM * tilesN;
  int row0 = bM * BM, col0 = bN * BN;
  int wm = (wv & 3) * 32;   // wave's M offset inside block tile
  int wn = (wv >> 2) * 64;  // wave's N offset inside block tile

  v8f acc[2][4];
#pragma unroll
  for (int i = 0; i < 2; ++i)
#pragma unroll
    for (int j = 0; j < 4; ++j) acc[i][j] = (v8f){};

  // per-thread staging chunks: 512 x 16B chunks per 128x32 bf16 panel,
  // 2 chunks per thread per panel
  int r0 = tid >> 2, s0 = (tid & 3) * 8;
  int r1 = (tid + 256) >> 2, s1 = ((tid + 256) & 3) * 8;
  uint32_t ldsA = (uint32_t)(uintptr_t)(void*)As;
  uint32_t ldsB = (uint32_t)(uintptr_t)(void*)Bs;
  uint32_t dA0 = ldsA + (uint32_t)(r0 * BK + s0) * 2;
  uint32_t dA1 = ldsA + (uint32_t)(r1 * BK + s1) * 2;
  uint32_t dB0 = ldsB + (uint32_t)(r0 * BK + s0) * 2;
  uint32_t dB1 = ldsB + (uint32_t)(r1 * BK + s1) * 2;
  const bf16* gA0 = A + (size_t)(row0 + r0) * Kp + s0;
  const bf16* gA1 = A + (size_t)(row0 + r1) * Kp + s1;
  const bf16* gB0 = Bt + (size_t)(col0 + r0) * Kp + s0;
  const bf16* gB1 = Bt + (size_t)(col0 + r1) * Kp + s1;

  for (int kk = 0; kk < Kp; kk += BK) {
    async_g2l_b128(dA0, gA0 + kk);
    async_g2l_b128(dA1, gA1 + kk);
    async_g2l_b128(dB0, gB0 + kk);
    async_g2l_b128(dB1, gB1 + kk);
    wait_async0();
    __syncthreads();
    v16bf af[2], bfr[4];
#pragma unroll
    for (int i = 0; i < 2; ++i) af[i] = load_fragA(As, BK, wm + i * 16, 0, lane);
#pragma unroll
    for (int j = 0; j < 4; ++j) bfr[j] = load_fragB(Bs, BK, wn + j * 16, 0, lane);
#pragma unroll
    for (int i = 0; i < 2; ++i)
#pragma unroll
      for (int j = 0; j < 4; ++j)
        acc[i][j] = wmma_bf16(af[i], bfr[j], acc[i][j]);
    __syncthreads();  // LDS reused next iteration
  }

  int n = lane & 15, hh = lane >> 4;
#pragma unroll
  for (int i = 0; i < 2; ++i)
#pragma unroll
    for (int j = 0; j < 4; ++j)
#pragma unroll
      for (int v = 0; v < 8; ++v)
        C[(size_t)(row0 + wm + i * 16 + hh * 8 + v) * N + col0 + wn + j * 16 + n] =
            acc[i][j][v];
}

// ---------------------------------------------------------------------------
// kv_prep: RMS-norm(kv) * w, Lorentz-project -> kv_pt(bf16, padded 544),
// and rope(k_pe) -> rk_pe (f32, 64 per token)
// ---------------------------------------------------------------------------
__global__ __launch_bounds__(256) void kv_prep(
    const float* __restrict__ kv_full, const float* __restrict__ kv_norm,
    const float* __restrict__ fcos, const float* __restrict__ fsin,
    bf16* __restrict__ kv_pt, float* __restrict__ rk_pe) {
  int s = blockIdx.x, t = threadIdx.x;
  int lane = t & 31, wid = t >> 5;
  __shared__ float red[8];
  const float* row = kv_full + (size_t)s * N_KVA_P;
  float v0 = row[t], v1 = row[t + 256];
  float ss = rsum32(v0 * v0 + v1 * v1);
  if (lane == 0) red[wid] = ss;
  __syncthreads();
  float tot = 0.0f;
#pragma unroll
  for (int i = 0; i < 8; ++i) tot += red[i];
  float rinv = 1.0f / sqrtf(tot * (1.0f / 512.0f) + 1e-6f);
  float n0 = v0 * rinv * kv_norm[t], n1 = v1 * rinv * kv_norm[t + 256];
  float ss2 = rsum32(n0 * n0 + n1 * n1);
  __syncthreads();
  if (lane == 0) red[wid] = ss2;
  __syncthreads();
  float tot2 = 0.0f;
#pragma unroll
  for (int i = 0; i < 8; ++i) tot2 += red[i];
  float tc = sqrtf(tot2 + 1.0f);               // time component, C = 1
  bf16* prow = kv_pt + (size_t)s * KP_KV;
  if (t == 0) prow[0] = f2bf(tc);
  prow[1 + t] = f2bf(n0);
  prow[1 + t + 256] = f2bf(n1);
  if (t < 31) prow[513 + t] = f2bf(0.0f);      // zero pad 513..543
  if (t < 32) {                                 // rope on k_pe (32 pairs)
    float x0 = row[KVRANK + 2 * t], x1 = row[KVRANK + 2 * t + 1];
    float c = fcos[(size_t)s * 32 + t], si = fsin[(size_t)s * 32 + t];
    rk_pe[(size_t)s * 64 + 2 * t] = x0 * c - x1 * si;
    rk_pe[(size_t)s * 64 + 2 * t + 1] = x0 * si + x1 * c;
  }
}

// ---------------------------------------------------------------------------
// q_prep: rope on q_pe, Lorentz-project, fold sign (-t), write bf16 (H,S,224)
// one wave per (s,h)
// ---------------------------------------------------------------------------
__global__ void q_prep(const float* __restrict__ q_raw,
                       const float* __restrict__ fcos,
                       const float* __restrict__ fsin, bf16* __restrict__ qs) {
  int h = blockIdx.x & (H_HEADS - 1);
  int s = blockIdx.x >> 4;
  int t = threadIdx.x;  // 0..31
  const float* qr = q_raw + (size_t)s * N_Q + h * 192;
  float nv[4];
  float ss = 0.0f;
#pragma unroll
  for (int j = 0; j < 4; ++j) { float v = qr[t * 4 + j]; nv[j] = v; ss += v * v; }
  float x0 = qr[128 + 2 * t], x1 = qr[129 + 2 * t];
  float c = fcos[(size_t)s * 32 + t], si = fsin[(size_t)s * 32 + t];
  float y0 = x0 * c - x1 * si, y1 = x0 * si + x1 * c;
  ss += y0 * y0 + y1 * y1;
  ss = rsum32(ss);
  float tc = sqrtf(ss + 1.0f);
  bf16* qrow = qs + ((size_t)h * S_LEN + s) * DQK_P;
  if (t == 0) qrow[0] = f2bf(-tc);             // signs = [-1, 1, ...] folded here
#pragma unroll
  for (int j = 0; j < 4; ++j) qrow[1 + t * 4 + j] = f2bf(nv[j]);
  qrow[1 + 128 + 2 * t] = f2bf(y0);
  qrow[1 + 129 + 2 * t] = f2bf(y1);
  if (t < 31) qrow[193 + t] = f2bf(0.0f);      // zero pad 193..223
}

// ---------------------------------------------------------------------------
// kv_assemble: k = [k_nope | rope(k_pe)] -> project -> ks (H,SPAD,224) bf16
//              v -> project -> vst transposed (H,128,SPAD) bf16
// one wave per (row,h); rows >= S are zero-filled slack
// ---------------------------------------------------------------------------
__global__ void kv_assemble(const float* __restrict__ kvb,
                            const float* __restrict__ rk_pe,
                            bf16* __restrict__ ks, bf16* __restrict__ vst) {
  int h = blockIdx.x & (H_HEADS - 1);
  int r = blockIdx.x >> 4;  // 0..SPAD-1
  int t = threadIdx.x;      // 0..31
  bf16* krow = ks + ((size_t)h * SPAD + r) * DQK_P;
  if (r >= S_LEN) {
#pragma unroll
    for (int j = 0; j < 7; ++j) krow[t * 7 + j] = f2bf(0.0f);
#pragma unroll
    for (int j = 0; j < 4; ++j)
      vst[((size_t)h * DV + t * 4 + j) * SPAD + r] = f2bf(0.0f);
    return;
  }
  const float* kb = kvb + (size_t)r * N_KVB_P + h * 255;
  float kd[6];
  float ss = 0.0f;
#pragma unroll
  for (int j = 0; j < 6; ++j) {
    int d = t * 6 + j;
    float v = (d < 128) ? kb[d] : rk_pe[(size_t)r * 64 + (d - 128)];
    kd[j] = v; ss += v * v;
  }
  ss = rsum32(ss);
  float tk = sqrtf(ss + 1.0f);
  if (t == 0) krow[0] = f2bf(tk);
#pragma unroll
  for (int j = 0; j < 6; ++j) krow[1 + t * 6 + j] = f2bf(kd[j]);
  if (t < 31) krow[193 + t] = f2bf(0.0f);
  // value side: vs = project(v), 128 dims, stored transposed
  float vv[4];
  float sv = 0.0f;
#pragma unroll
  for (int j = 0; j < 4; ++j) {
    int vi = t * 4 + j;
    float v = (vi == 0) ? 0.0f : kb[128 + vi - 1];
    vv[j] = v; sv += v * v;
  }
  sv = rsum32(sv);
  if (t == 0) vv[0] = sqrtf(sv + 1.0f);
#pragma unroll
  for (int j = 0; j < 4; ++j)
    vst[((size_t)h * DV + t * 4 + j) * SPAD + r] = f2bf(vv[j]);
}

// ---------------------------------------------------------------------------
// flash attention: one wave owns 16 query rows; streams causal key tiles of
// 32 with online softmax.  QK^T: 7 WMMAs x2 subtiles; P*V: 8 WMMAs.
// P re-swizzle D-layout -> A-layout goes through per-wave LDS (s_wait_dscnt),
// next K/V tiles are prefetched (global_prefetch_b8).
// ---------------------------------------------------------------------------
__global__ __launch_bounds__(256, 1) void attn_fwd(
    const bf16* __restrict__ qs, const bf16* __restrict__ ks,
    const bf16* __restrict__ vst, float* __restrict__ ave,
    const float* __restrict__ sscale, const float* __restrict__ biasp) {
  __shared__ __align__(32) bf16 lds_p[8][16 * 32];
  int lane = threadIdx.x & 31;
  int wv = threadIdx.x >> 5;
  int gw = blockIdx.x * 8 + wv;
  int qt = gw & ((S_LEN / 16) - 1);
  int h = gw / (S_LEN / 16);
  int q0 = qt * 16;
  int m = lane & 15, hh = lane >> 4;
  const bf16* qsh = qs + (size_t)h * S_LEN * DQK_P;
  const bf16* ksh = ks + (size_t)h * SPAD * DQK_P;
  const bf16* vh = vst + (size_t)h * DV * SPAD;
  float inv_scale = 1.0f / sscale[0];
  float scal = 2.0f * inv_scale;                  // scores = (2C + 2*dot)/scale + bias
  float sadd = 2.0f * inv_scale + biasp[0];       // C == 1

  v16bf qa[7];
#pragma unroll
  for (int t = 0; t < 7; ++t) qa[t] = load_fragA(qsh, DQK_P, q0, t * 32, lane);

  v8f acc[8];
#pragma unroll
  for (int i = 0; i < 8; ++i) acc[i] = (v8f){};
  float mr[8], lr[8];
#pragma unroll
  for (int v = 0; v < 8; ++v) { mr[v] = -1e30f; lr[v] = 0.0f; }

  for (int k0 = 0; k0 <= q0 + 15; k0 += 32) {
    // prefetch next key/value tiles into cache while this tile computes
    __builtin_prefetch((const void*)(ksh + (size_t)(k0 + 32 + m) * DQK_P), 0, 1);
    __builtin_prefetch((const void*)(vh + (size_t)(m * 16) * SPAD + k0 + 32), 0, 1);
    v8f c0 = {}, c1 = {};
#pragma unroll
    for (int t = 0; t < 7; ++t) {
      v16bf b0 = load_fragB(ksh, DQK_P, k0, t * 32, lane);
      v16bf b1 = load_fragB(ksh, DQK_P, k0 + 16, t * 32, lane);
      c0 = wmma_bf16(qa[t], b0, c0);
      c1 = wmma_bf16(qa[t], b1, c1);
    }
    float p0[8], p1[8], alpha[8];
#pragma unroll
    for (int v = 0; v < 8; ++v) {
      int qi = q0 + hh * 8 + v;
      int kj0 = k0 + m, kj1 = kj0 + 16;
      float s0 = (kj0 <= qi) ? (scal * c0[v] + sadd) : -1e30f;
      float s1 = (kj1 <= qi) ? (scal * c1[v] + sadd) : -1e30f;
      float mx = rmax16(fmaxf(s0, s1));
      float mn = fmaxf(mr[v], mx);
      alpha[v] = __expf(mr[v] - mn);
      float e0 = __expf(s0 - mn);
      float e1 = __expf(s1 - mn);
      float rs = rsum16(e0 + e1);
      lr[v] = lr[v] * alpha[v] + rs;
      mr[v] = mn;
      p0[v] = e0; p1[v] = e1;
    }
#pragma unroll
    for (int nt = 0; nt < 8; ++nt)
#pragma unroll
      for (int v = 0; v < 8; ++v) acc[nt][v] *= alpha[v];
    // stage P (D-layout) into LDS, re-read in A-layout
    bf16* lp = lds_p[wv];
#pragma unroll
    for (int v = 0; v < 8; ++v) {
      lp[(hh * 8 + v) * 32 + m] = f2bf(p0[v]);
      lp[(hh * 8 + v) * 32 + 16 + m] = f2bf(p1[v]);
    }
    asm volatile("s_wait_dscnt 0" ::: "memory");
    v16bf pa;
    {
      v8bf lo = *(const v8bf*)&lp[m * 32 + hh * 8];
      v8bf hi = *(const v8bf*)&lp[m * 32 + 16 + hh * 8];
#pragma unroll
      for (int i = 0; i < 8; ++i) { pa[i] = lo[i]; pa[i + 8] = hi[i]; }
    }
#pragma unroll
    for (int nt = 0; nt < 8; ++nt) {
      v16bf vb = load_fragB(vh, SPAD, nt * 16, k0, lane);
      acc[nt] = wmma_bf16(pa, vb, acc[nt]);
    }
  }
#pragma unroll
  for (int nt = 0; nt < 8; ++nt)
#pragma unroll
    for (int v = 0; v < 8; ++v)
      ave[((size_t)h * S_LEN + q0 + hh * 8 + v) * DV + nt * 16 + m] =
          acc[nt][v] / lr[v];
}

// ---------------------------------------------------------------------------
// Lorentz centroid normalization + head transpose -> bf16 (S, 2048)
// ---------------------------------------------------------------------------
__global__ void centroid(const float* __restrict__ ave, bf16* __restrict__ aout) {
  int h = blockIdx.x & (H_HEADS - 1);
  int s = blockIdx.x >> 4;
  int t = threadIdx.x;  // 0..31
  const float* a = ave + ((size_t)h * S_LEN + s) * DV;
  float av[4];
  float ss = 0.0f;
#pragma unroll
  for (int j = 0; j < 4; ++j) { av[j] = a[t * 4 + j]; ss += av[j] * av[j]; }
  ss = rsum32(ss);
  float a0 = __shfl(av[0], 0, 32);
  float inner = ss - 2.0f * a0 * a0;             // -a0^2 + sum_{i>=1} a_i^2
  float rinv = 1.0f / sqrtf(fmaxf(fabsf(inner), 1e-8f));
  bf16* orow = aout + (size_t)s * 2048 + h * DV;
#pragma unroll
  for (int j = 0; j < 4; ++j) orow[t * 4 + j] = f2bf(av[j] * rinv);
}

// ---------------------------------------------------------------------------
// final projection: out = [sqrt(|o|^2 + 1), o]
// ---------------------------------------------------------------------------
__global__ __launch_bounds__(256) void final_project(const float* __restrict__ o,
                                                     float* __restrict__ out) {
  int s = blockIdx.x, t = threadIdx.x;
  int lane = t & 31, wid = t >> 5;
  __shared__ float red[8];
  const float* row = o + (size_t)s * 2048;
  float ss = 0.0f;
  for (int j = t; j < 2047; j += 256) { float v = row[j]; ss += v * v; }
  ss = rsum32(ss);
  if (lane == 0) red[wid] = ss;
  __syncthreads();
  float tot = 0.0f;
#pragma unroll
  for (int i = 0; i < 8; ++i) tot += red[i];
  float tc = sqrtf(tot + 1.0f);
  float* orow = out + (size_t)s * 2048;
  if (t == 0) orow[0] = tc;
  for (int j = t; j < 2047; j += 256) orow[1 + j] = row[j];
}

// ---------------------------------------------------------------------------
// workspace layout (lifetime unions)
// ---------------------------------------------------------------------------
static constexpr size_t AL(size_t x) { return (x + 255) & ~size_t(255); }
static constexpr size_t SZ_XB = AL((size_t)S_LEN * DIM_ * 2);
static constexpr size_t SZ_WQT = AL((size_t)N_Q * DIM_ * 2);
static constexpr size_t SZ_WKVBT = AL((size_t)N_KVB_P * KP_KV * 2);
static constexpr size_t SZ_WOT = AL((size_t)2048 * 2048 * 2);
static constexpr size_t SZ_WU =
    (SZ_WQT > SZ_WKVBT + SZ_WOT) ? SZ_WQT : (SZ_WKVBT + SZ_WOT);
static constexpr size_t SZ_WKVAT = AL((size_t)N_KVA_P * DIM_ * 2);
static constexpr size_t SZ_QRAW = AL((size_t)S_LEN * N_Q * 4);
static constexpr size_t SZ_AOUT = AL((size_t)S_LEN * 2048 * 2);
static constexpr size_t SZ_O = AL((size_t)S_LEN * 2048 * 4);
static constexpr size_t SZ_U2 =
    (SZ_QRAW > SZ_AOUT + SZ_O) ? SZ_QRAW : (SZ_AOUT + SZ_O);
static constexpr size_t SZ_KVF = AL((size_t)S_LEN * N_KVA_P * 4);
static constexpr size_t SZ_KVPT = AL((size_t)S_LEN * KP_KV * 2);
static constexpr size_t SZ_RKPE = AL((size_t)S_LEN * 64 * 4);
static constexpr size_t SZ_KVB = AL((size_t)S_LEN * N_KVB_P * 4);
static constexpr size_t SZ_QS = AL((size_t)H_HEADS * S_LEN * DQK_P * 2);
static constexpr size_t SZ_KS = AL((size_t)H_HEADS * SPAD * DQK_P * 2);
static constexpr size_t SZ_VST = AL((size_t)H_HEADS * DV * SPAD * 2);

static constexpr size_t OFF_XB = 0;
static constexpr size_t OFF_WU = OFF_XB + SZ_XB;
static constexpr size_t OFF_WQT = OFF_WU;        // dead after GEMM1
static constexpr size_t OFF_WKVBT = OFF_WU;      // reuses wq_t region
static constexpr size_t OFF_WOT = OFF_WU + SZ_WKVBT;
static constexpr size_t OFF_WKVAT = OFF_WU + SZ_WU;
static constexpr size_t OFF_U2 = OFF_WKVAT + SZ_WKVAT;
static constexpr size_t OFF_QRAW = OFF_U2;       // dead after q_prep
static constexpr size_t OFF_AOUT = OFF_U2;       // reuses q_raw region
static constexpr size_t OFF_O = OFF_U2 + SZ_AOUT;
static constexpr size_t OFF_KVF = OFF_U2 + SZ_U2;
static constexpr size_t OFF_KVPT = OFF_KVF + SZ_KVF;
static constexpr size_t OFF_RKPE = OFF_KVPT + SZ_KVPT;
static constexpr size_t OFF_KVB = OFF_RKPE + SZ_RKPE;
static constexpr size_t OFF_QS = OFF_KVB + SZ_KVB;
static constexpr size_t OFF_KS = OFF_QS + SZ_QS;
static constexpr size_t OFF_VST = OFF_KS + SZ_KS;
static constexpr size_t OFF_AVE = OFF_VST + SZ_VST;

extern "C" void kernel_launch(void* const* d_in, const int* in_sizes, int n_in,
                              void* d_out, int out_size, void* d_ws,
                              size_t ws_size, hipStream_t stream) {
  (void)in_sizes; (void)n_in; (void)out_size; (void)ws_size;
  const float* x = (const float*)d_in[0];
  const float* fcos = (const float*)d_in[2];
  const float* fsin = (const float*)d_in[3];
  const float* wq = (const float*)d_in[5];
  const float* wkv_a = (const float*)d_in[6];
  const float* kvnorm = (const float*)d_in[7];
  const float* wkv_b = (const float*)d_in[8];
  const float* wo = (const float*)d_in[9];
  const float* sscale = (const float*)d_in[10];
  const float* biasp = (const float*)d_in[11];
  float* out = (float*)d_out;
  char* ws = (char*)d_ws;

  bf16* xb = (bf16*)(ws + OFF_XB);
  bf16* wq_t = (bf16*)(ws + OFF_WQT);
  bf16* wkvb_t = (bf16*)(ws + OFF_WKVBT);
  bf16* wo_t = (bf16*)(ws + OFF_WOT);
  bf16* wkva_t = (bf16*)(ws + OFF_WKVAT);
  float* q_raw = (float*)(ws + OFF_QRAW);
  bf16* aout = (bf16*)(ws + OFF_AOUT);
  float* obuf = (float*)(ws + OFF_O);
  float* kv_full = (float*)(ws + OFF_KVF);
  bf16* kv_pt = (bf16*)(ws + OFF_KVPT);
  float* rk_pe = (float*)(ws + OFF_RKPE);
  float* kvb = (float*)(ws + OFF_KVB);
  bf16* qs = (bf16*)(ws + OFF_QS);
  bf16* ksb = (bf16*)(ws + OFF_KS);
  bf16* vst = (bf16*)(ws + OFF_VST);
  float* ave = (float*)(ws + OFF_AVE);

  auto nb = [](size_t total) { return (unsigned)((total + 255) / 256); };
  auto gb = [](int M, int N) { return (unsigned)((M / BM) * (N / BN)); };

  // 1. x -> bf16 ; wq -> bf16^T ; q_raw = x @ wq
  cvt_pad<<<nb((size_t)S_LEN * DIM_), 256, 0, stream>>>(x, xb, S_LEN, DIM_, DIM_);
  cvt_tpad<<<nb((size_t)N_Q * DIM_), 256, 0, stream>>>(wq, wq_t, DIM_, N_Q, N_Q, DIM_);
  gemm_bf16_kernel<<<gb(S_LEN, N_Q), 256, 0, stream>>>(xb, wq_t, q_raw, S_LEN, N_Q, DIM_);
  // 2. kv_full = x @ wkv_a   (N padded 576 -> 640)
  cvt_tpad<<<nb((size_t)N_KVA_P * DIM_), 256, 0, stream>>>(wkv_a, wkva_t, DIM_, N_KVA, N_KVA_P, DIM_);
  gemm_bf16_kernel<<<gb(S_LEN, N_KVA_P), 256, 0, stream>>>(xb, wkva_t, kv_full, S_LEN, N_KVA_P, DIM_);
  // 3. rmsnorm + project + rope(k_pe)
  kv_prep<<<S_LEN, 256, 0, stream>>>(kv_full, kvnorm, fcos, fsin, kv_pt, rk_pe);
  // 4. kvb = kv_pt @ wkv_b   (N padded 4080 -> 4096)
  cvt_tpad<<<nb((size_t)N_KVB_P * KP_KV), 256, 0, stream>>>(wkv_b, wkvb_t, 513, N_KVB, N_KVB_P, KP_KV);
  gemm_bf16_kernel<<<gb(S_LEN, N_KVB_P), 256, 0, stream>>>(kv_pt, wkvb_t, kvb, S_LEN, N_KVB_P, KP_KV);
  // 5. q/k/v assembly (rope, project, sign fold, transpose V)
  q_prep<<<S_LEN * H_HEADS, 32, 0, stream>>>(q_raw, fcos, fsin, qs);
  kv_assemble<<<SPAD * H_HEADS, 32, 0, stream>>>(kvb, rk_pe, ksb, vst);
  // 6. flash attention (WMMA QK^T + softmax + WMMA P*V)
  attn_fwd<<<(H_HEADS * (S_LEN / 16)) / 8, 256, 0, stream>>>(qs, ksb, vst, ave, sscale, biasp);
  // 7. centroid normalize + transpose back
  centroid<<<S_LEN * H_HEADS, 32, 0, stream>>>(ave, aout);
  // 8. o = aout @ wo ; final Lorentz projection
  cvt_tpad<<<nb((size_t)2048 * 2048), 256, 0, stream>>>(wo, wo_t, 2048, 2047, 2048, 2048);
  gemm_bf16_kernel<<<gb(2048, 2048), 256, 0, stream>>>(aout, wo_t, obuf, 2048, 2048, 2048);
  final_project<<<S_LEN, 256, 0, stream>>>(obuf, out);
}
